// RelPosTransformerLayer_25374666784997
// MI455X (gfx1250) — compile-verified, matrix operand order checked by reading, not run on hardware
//
#include <hip/hip_runtime.h>
#include <hip/hip_bf16.h>

// ---------------- types ----------------
typedef __bf16 v16bf __attribute__((ext_vector_type(16)));
typedef __bf16 v8bf  __attribute__((ext_vector_type(8)));
typedef float  v8f   __attribute__((ext_vector_type(8)));
typedef int    v4i   __attribute__((vector_size(16)));     // matches builtin param

#define NB   4
#define LSEQ 1024
#define DMOD 768
#define FMOD 3072
#define NH   12
#define DK   64
#define TBL  201
#define TBLN (TBL*TBL)            // 40401 floats = 161604 B
#define TBL_BYTES_AL 161616       // aligned to 16

// ---- async global->LDS DMA (CDNA5), guarded: falls back to sync copy ----
#if defined(__AMDGCN__) && __has_builtin(__builtin_amdgcn_global_load_async_to_lds_b128)
#define USE_ASYNC_LDS 1
#else
#define USE_ASYNC_LDS 0
#endif

static __device__ inline void cp16_to_lds(void* lds, const void* g) {
#if USE_ASYNC_LDS
  // integer casts give: global flat addr == as(1) addr; LDS flat addr low 32b == LDS offset
  auto gp = (__attribute__((address_space(1))) v4i*)(size_t)g;
  auto lp = (__attribute__((address_space(3))) v4i*)(unsigned)(size_t)lds;
  __builtin_amdgcn_global_load_async_to_lds_b128(gp, lp, 0, 0);
#else
  *(uint4*)lds = *(const uint4*)g;
#endif
}

static __device__ inline void async_wait0() {
#if USE_ASYNC_LDS
#if __has_builtin(__builtin_amdgcn_s_wait_asynccnt)
  __builtin_amdgcn_s_wait_asynccnt(0);
#else
  asm volatile("s_wait_asynccnt 0x0" ::: "memory");
#endif
#endif
}

static __device__ inline __bf16 f2bf(float f) {
  unsigned int u = __builtin_bit_cast(unsigned int, f);
  u += 0x7FFFu + ((u >> 16) & 1u);          // round-to-nearest-even
  unsigned short s = (unsigned short)(u >> 16);
  return __builtin_bit_cast(__bf16, s);
}

// build a bf16 fragment (A 16x32 or B 32x16) from two 16B-aligned 8-elem runs
static __device__ inline v16bf load_frag(const __bf16* p0, const __bf16* p1) {
  v8bf lo = *(const v8bf*)p0;
  v8bf hi = *(const v8bf*)p1;
  return __builtin_shufflevector(lo, hi, 0,1,2,3,4,5,6,7,8,9,10,11,12,13,14,15);
}

static __device__ inline v8f wmma_bf16(v16bf a, v16bf b, v8f c) {
  return __builtin_amdgcn_wmma_f32_16x16x32_bf16(false, a, false, b, (short)0, c, false, false);
}

// ---------------- f32 -> bf16 convert ----------------
__global__ void cvt_kernel(const float* __restrict__ in, __bf16* __restrict__ out, int n) {
  int i = blockIdx.x * 256 + threadIdx.x;
  if (i < n) out[i] = f2bf(in[i]);
}

// ---------------- tiled WMMA GEMM, double-buffered async LDS staging ----------------
// C[M,N] = A[M,K](bf16) * B[K,N](bf16) + bias[N], optional relu.
// MODE 0: f32 row-major   1: bf16 row-major
// MODE 2: bf16 [B,H,L,dk] (heads, for Q/K)   3: bf16 [B,H,dk,L] (V transposed)
template<int MODE>
__global__ __launch_bounds__(256)
void gemm_kernel(const __bf16* __restrict__ A, const __bf16* __restrict__ Bm,
                 const float* __restrict__ bias, void* __restrict__ Cout,
                 int Mdim, int Ndim, int Kdim, int relu)
{
  __shared__ alignas(16) __bf16 As[2][128 * 40];  // 128 x 32 tiles, stride 40
  __shared__ alignas(16) __bf16 Bs[2][64 * 40];   // transposed: Bs[n][k]
  const int tid = threadIdx.x;
  const int w = tid >> 5, lane = tid & 31;
  const int lan15 = lane & 15, khalf = lane >> 4;
  const int wm = w >> 1, wn = w & 1;              // 4x2 wave grid, 32x32 per wave
  const int m0 = blockIdx.y * 128, n0 = blockIdx.x * 64;

  // per-thread staging assignments
  const int arow = tid >> 1, aseg = (tid & 1) * 16;          // A: 16 bf16 per thread
  const int bk = tid >> 3, bn = (tid & 7) * 8;               // B: 8 bf16 per thread
  const __bf16* Ag = A + (size_t)(m0 + arow) * Kdim + aseg;
  const __bf16* Bg = Bm + (size_t)bk * Ndim + n0 + bn;

  v8f acc[2][2] = {};
  uint4 breg;

  auto stageA = [&](int k0, int buf) {
    __bf16* dst = &As[buf][arow * 40 + aseg];
    cp16_to_lds(dst,     Ag + k0);
    cp16_to_lds(dst + 8, Ag + k0 + 8);
  };
  auto loadB = [&](int k0) { breg = *(const uint4*)(Bg + (size_t)k0 * Ndim); };
  auto storeB = [&](int buf) {
    const __bf16* tv = (const __bf16*)&breg;
    #pragma unroll
    for (int j = 0; j < 8; ++j) Bs[buf][(bn + j) * 40 + bk] = tv[j];
  };

  // prologue: stage tile 0
  stageA(0, 0);
  loadB(0);
  storeB(0);
  async_wait0();
  __syncthreads();

  int buf = 0;
  for (int k0 = 0; k0 < Kdim; k0 += 32) {
    const int nxt = buf ^ 1;
    const bool more = (k0 + 32) < Kdim;
    if (more) { stageA(k0 + 32, nxt); loadB(k0 + 32); }   // overlap DMA with WMMA

    v16bf af[2], bf[2];
    #pragma unroll
    for (int i = 0; i < 2; ++i) {
      const __bf16* p = &As[buf][(wm * 32 + i * 16 + lan15) * 40];
      af[i] = load_frag(p + khalf * 8, p + 16 + khalf * 8);
    }
    #pragma unroll
    for (int j = 0; j < 2; ++j) {
      const __bf16* p = &Bs[buf][(wn * 32 + j * 16 + lan15) * 40];
      bf[j] = load_frag(p + khalf * 8, p + 16 + khalf * 8);
    }
    #pragma unroll
    for (int i = 0; i < 2; ++i)
      #pragma unroll
      for (int j = 0; j < 2; ++j)
        acc[i][j] = wmma_bf16(af[i], bf[j], acc[i][j]);

    if (more) storeB(nxt);
    async_wait0();
    __syncthreads();
    buf = nxt;
  }

  // epilogue: C/D layout — lane = column (lan15), VGPR r = row r + 8*khalf
  #pragma unroll
  for (int i = 0; i < 2; ++i)
    #pragma unroll
    for (int j = 0; j < 2; ++j)
      #pragma unroll
      for (int r = 0; r < 8; ++r) {
        int gm = m0 + wm * 32 + i * 16 + r + (khalf << 3);
        int gn = n0 + wn * 32 + j * 16 + lan15;
        float v = acc[i][j][r] + bias[gn];
        if (relu) v = fmaxf(v, 0.0f);
        if (MODE == 0) {
          ((float*)Cout)[(size_t)gm * Ndim + gn] = v;
        } else if (MODE == 1) {
          ((__bf16*)Cout)[(size_t)gm * Ndim + gn] = f2bf(v);
        } else {
          int bb = gm >> 10, l = gm & 1023, hh = gn >> 6, dc = gn & 63;
          size_t off = (MODE == 2)
              ? (((size_t)(bb * NH + hh) * LSEQ + l) * DK + dc)     // [B,H,L,dk]
              : (((size_t)(bb * NH + hh) * DK + dc) * LSEQ + l);    // [B,H,dk,L]
          ((__bf16*)Cout)[off] = f2bf(v);
        }
      }
}

// ---------------- flash attention with rel-pos bias ----------------
// grid (L/128, B*H), 256 threads = 8 waves; wave w handles 16 query rows.
// Dynamic LDS: per-head bias table (161.6 KB) + 8 per-wave P bounce tiles.
__global__ __launch_bounds__(256)
void attn_kernel(const __bf16* __restrict__ Qb, const __bf16* __restrict__ Kb,
                 const __bf16* __restrict__ Vb, const float* __restrict__ pos,
                 const float* __restrict__ pos_mask, const float* __restrict__ att_mask,
                 const unsigned char* __restrict__ padmask,
                 const float* __restrict__ table_g, __bf16* __restrict__ ctxb)
{
  extern __shared__ char smem[];
  float*  tbl = (float*)smem;
  __bf16* PsA = (__bf16*)(smem + TBL_BYTES_AL);

  const int tid = threadIdx.x;
  const int bh = blockIdx.y;
  const int b = bh / NH, h = bh % NH;

  // stage this head's 201x201 bias table into LDS (fits in 320KB WGP LDS)
  const float* tg = table_g + (size_t)h * TBLN;
  for (int i = tid; i < TBLN; i += 256) tbl[i] = tg[i];
  __syncthreads();

  const int w = tid >> 5, lane = tid & 31;
  const int lan15 = lane & 15, khalf = lane >> 4;
  __bf16* Ps = PsA + w * (16 * 40);          // 16 rows x 32 keys, stride 40

  const int qbase = blockIdx.x * 128 + w * 16;
  const size_t bL = (size_t)b * LSEQ;

  // Q fragments (16 x 64, two K=32 chunks), [B,H,L,dk] row-major
  const __bf16* Qrow = Qb + ((size_t)bh * LSEQ + qbase + lan15) * DK;
  v16bf qf[2];
  qf[0] = load_frag(Qrow + khalf * 8,      Qrow + 16 + khalf * 8);
  qf[1] = load_frag(Qrow + 32 + khalf * 8, Qrow + 48 + khalf * 8);

  // per-lane query-row attributes (lane l holds row l&15), then broadcast per
  // accumulator row ONCE (loop-invariant across all key tiles)
  const int qrow_l = qbase + lan15;
  const float qx  = pos[(bL + qrow_l) * 2 + 0];
  const float qy  = pos[(bL + qrow_l) * 2 + 1];
  const float pmq = pos_mask[bL + qrow_l];

  float qxr[8], qyr[8], pmqr[8];
  const float* amrow[8];
  #pragma unroll
  for (int r = 0; r < 8; ++r) {
    const int qm = r + (khalf << 3);
    qxr[r]  = __shfl(qx,  qm, 32);
    qyr[r]  = __shfl(qy,  qm, 32);
    pmqr[r] = __shfl(pmq, qm, 32);
    amrow[r] = att_mask + (bL + qbase + qm) * LSEQ;
  }

  float mrun[8], lrun[8];
  v8f ctx[4] = {};
  #pragma unroll
  for (int r = 0; r < 8; ++r) { mrun[r] = -3.0e38f; lrun[r] = 0.0f; }

  const __bf16* Kbase = Kb + (size_t)bh * LSEQ * DK;
  const __bf16* Vbase = Vb + (size_t)bh * DK * LSEQ;

  for (int kt = 0; kt < LSEQ / 32; ++kt) {
    const int key0 = kt * 32;

    if (kt + 1 < LSEQ / 32) {      // pull next K/V tiles toward the WGP
      __builtin_prefetch(Kbase + (size_t)(key0 + 32 + lan15) * DK, 0, 0);
      __builtin_prefetch(Vbase + (size_t)lan15 * LSEQ + key0 + 32, 0, 0);
    }

    v8f sc[2];
    // scores: Q(16x64) x K^T(64x32) -> two 16x16 tiles, chained over K
    #pragma unroll
    for (int s = 0; s < 2; ++s) {
      const __bf16* Kr = Kbase + (size_t)(key0 + s * 16 + lan15) * DK;
      v16bf kf0 = load_frag(Kr + khalf * 8,      Kr + 16 + khalf * 8);
      v16bf kf1 = load_frag(Kr + 32 + khalf * 8, Kr + 48 + khalf * 8);
      v8f z = {};
      z = wmma_bf16(qf[0], kf0, z);
      z = wmma_bf16(qf[1], kf1, z);
      sc[s] = z;
    }

    // scale + rel-pos bias gather (LDS) + masking
    #pragma unroll
    for (int s = 0; s < 2; ++s) {
      const int key = key0 + s * 16 + lan15;       // lane = score column = key
      const float kx  = pos[(bL + key) * 2 + 0];
      const float ky  = pos[(bL + key) * 2 + 1];
      const float pmk = pos_mask[bL + key];
      const bool  pad = padmask[bL + key] != 0;
      #pragma unroll
      for (int r = 0; r < 8; ++r) {
        // rintf == round-half-even == jnp.round, lowers to v_rndne_f32
        float dxf = fminf(fmaxf(rintf(qxr[r] - kx), -100.0f), 100.0f) + 100.0f;
        float dyf = fminf(fmaxf(rintf(qyr[r] - ky), -100.0f), 100.0f) + 100.0f;
        int idx = (int)dyf * TBL + (int)dxf;
        float bias = tbl[idx] * (pmqr[r] * pmk);
        float am = amrow[r][key];
        float v = sc[s][r] * 0.125f + bias;        // 1/sqrt(64)
        if (am == 0.0f || pad) v = -1.0e9f;
        sc[s][r] = v;
      }
    }

    // online softmax: row reductions across the 16 lanes of each half
    float alpha[8];
    #pragma unroll
    for (int r = 0; r < 8; ++r) {
      float t = fmaxf(sc[0][r], sc[1][r]);
      t = fmaxf(t, __shfl_xor(t, 1, 32));
      t = fmaxf(t, __shfl_xor(t, 2, 32));
      t = fmaxf(t, __shfl_xor(t, 4, 32));
      t = fmaxf(t, __shfl_xor(t, 8, 32));
      float mnew = fmaxf(mrun[r], t);
      float a = __expf(mrun[r] - mnew);
      mrun[r] = mnew;
      float p0 = __expf(sc[0][r] - mnew);
      float p1 = __expf(sc[1][r] - mnew);
      float sm = p0 + p1;
      sm += __shfl_xor(sm, 1, 32);
      sm += __shfl_xor(sm, 2, 32);
      sm += __shfl_xor(sm, 4, 32);
      sm += __shfl_xor(sm, 8, 32);
      lrun[r] = lrun[r] * a + sm;
      alpha[r] = a;
      sc[0][r] = p0;
      sc[1][r] = p1;
    }
    #pragma unroll
    for (int j = 0; j < 4; ++j)
      #pragma unroll
      for (int r = 0; r < 8; ++r) ctx[j][r] *= alpha[r];

    // transpose P (C-layout) -> A-layout via per-wave LDS bounce (in-order DS)
    #pragma unroll
    for (int s = 0; s < 2; ++s)
      #pragma unroll
      for (int r = 0; r < 8; ++r) {
        int qm = r + (khalf << 3);
        Ps[qm * 40 + s * 16 + lan15] = f2bf(sc[s][r]);
      }
    const __bf16* Pr = Ps + lan15 * 40;
    v16bf pf = load_frag(Pr + khalf * 8, Pr + 16 + khalf * 8);

    // ctx(16x64) += P(16x32) x V(32x64); V is [dk][L] so columns are contiguous
    #pragma unroll
    for (int j = 0; j < 4; ++j) {
      const __bf16* Vr = Vbase + (size_t)(j * 16 + lan15) * LSEQ + key0;
      v16bf vf = load_frag(Vr + khalf * 8, Vr + 16 + khalf * 8);
      ctx[j] = wmma_bf16(pf, vf, ctx[j]);
    }
  }

  // normalize and write ctx back as [B, L, H*dk] bf16
  #pragma unroll
  for (int r = 0; r < 8; ++r) {
    float inv = 1.0f / fmaxf(lrun[r], 1e-20f);
    int qm = r + (khalf << 3);
    size_t rowoff = (bL + qbase + qm) * DMOD + h * DK;
    #pragma unroll
    for (int j = 0; j < 4; ++j)
      ctxb[rowoff + j * 16 + lan15] = f2bf(ctx[j][r] * inv);
  }
}

// ---------------- residual + LayerNorm (D=768, 256 thr, 3 elems/thr) ----------------
__global__ __launch_bounds__(256)
void ln_kernel(const float* __restrict__ x, const float* __restrict__ delta,
               const float* __restrict__ g, const float* __restrict__ bta,
               float* __restrict__ outf, __bf16* __restrict__ outb)
{
  __shared__ float red[256];
  const int row = blockIdx.x, tid = threadIdx.x;
  const float* xr = x + (size_t)row * DMOD;
  const float* dr = delta + (size_t)row * DMOD;
  float vals[3], s = 0.0f, ss = 0.0f;
  #pragma unroll
  for (int i = 0; i < 3; ++i) {
    int c = tid + i * 256;
    float v = xr[c] + dr[c];
    vals[i] = v; s += v; ss += v * v;
  }
  red[tid] = s; __syncthreads();
  for (int off = 128; off > 0; off >>= 1) { if (tid < off) red[tid] += red[tid + off]; __syncthreads(); }
  float mean = red[0] * (1.0f / DMOD);
  __syncthreads();
  red[tid] = ss; __syncthreads();
  for (int off = 128; off > 0; off >>= 1) { if (tid < off) red[tid] += red[tid + off]; __syncthreads(); }
  float var = red[0] * (1.0f / DMOD) - mean * mean;
  float rstd = rsqrtf(var + 1e-5f);
  #pragma unroll
  for (int i = 0; i < 3; ++i) {
    int c = tid + i * 256;
    float y = (vals[i] - mean) * rstd * g[c] + bta[c];
    outf[(size_t)row * DMOD + c] = y;
    if (outb) outb[(size_t)row * DMOD + c] = f2bf(y);
  }
}

// ---------------- host orchestration ----------------
extern "C" void kernel_launch(void* const* d_in, const int* in_sizes, int n_in,
                              void* d_out, int out_size, void* d_ws, size_t ws_size,
                              hipStream_t stream) {
  (void)in_sizes; (void)n_in; (void)out_size; (void)ws_size;
  const float* tokens   = (const float*)d_in[0];
  const float* pos      = (const float*)d_in[1];
  const float* pos_mask = (const float*)d_in[2];
  const float* att_mask = (const float*)d_in[3];
  const unsigned char* padding = (const unsigned char*)d_in[4];
  const float* Wq = (const float*)d_in[5];  const float* bq = (const float*)d_in[6];
  const float* Wk = (const float*)d_in[7];  const float* bk = (const float*)d_in[8];
  const float* Wv = (const float*)d_in[9];  const float* bv = (const float*)d_in[10];
  const float* Wo = (const float*)d_in[11]; const float* bo = (const float*)d_in[12];
  const float* ptab = (const float*)d_in[13];
  const float* W1 = (const float*)d_in[14]; const float* b1 = (const float*)d_in[15];
  const float* W2 = (const float*)d_in[16]; const float* b2 = (const float*)d_in[17];
  const float* ln1g = (const float*)d_in[18]; const float* ln1b = (const float*)d_in[19];
  const float* ln2g = (const float*)d_in[20]; const float* ln2b = (const float*)d_in[21];

  const int BL = NB * LSEQ;                       // 4096
  char* p = (char*)d_ws;
  auto alloc = [&](size_t bytes) -> char* {
    char* r = p; p += (bytes + 255) & ~(size_t)255; return r;
  };
  __bf16* Xbf  = (__bf16*)alloc((size_t)BL * DMOD * 2);
  __bf16* Wqb  = (__bf16*)alloc((size_t)DMOD * DMOD * 2);
  __bf16* Wkb  = (__bf16*)alloc((size_t)DMOD * DMOD * 2);
  __bf16* Wvb  = (__bf16*)alloc((size_t)DMOD * DMOD * 2);
  __bf16* Wob  = (__bf16*)alloc((size_t)DMOD * DMOD * 2);
  __bf16* W1b  = (__bf16*)alloc((size_t)DMOD * FMOD * 2);
  __bf16* W2b  = (__bf16*)alloc((size_t)FMOD * DMOD * 2);
  __bf16* Qb   = (__bf16*)alloc((size_t)BL * DMOD * 2);    // [B,H,L,dk]
  __bf16* Kbuf = (__bf16*)alloc((size_t)BL * DMOD * 2);    // [B,H,L,dk]
  __bf16* Vbuf = (__bf16*)alloc((size_t)BL * DMOD * 2);    // [B,H,dk,L]
  __bf16* ctxb = (__bf16*)alloc((size_t)BL * DMOD * 2);    // [B,L,D]
  float*  aout = (float*) alloc((size_t)BL * DMOD * 4);
  float*  tok1 = (float*) alloc((size_t)BL * DMOD * 4);
  __bf16* tok1b= (__bf16*)alloc((size_t)BL * DMOD * 2);
  __bf16* ff1b = (__bf16*)alloc((size_t)BL * FMOD * 2);
  float*  ff2  = (float*) alloc((size_t)BL * DMOD * 4);

  auto cvt = [&](const float* src, __bf16* dst, int n) {
    cvt_kernel<<<(n + 255) / 256, 256, 0, stream>>>(src, dst, n);
  };
  cvt(tokens, Xbf, BL * DMOD);
  cvt(Wq, Wqb, DMOD * DMOD);  cvt(Wk, Wkb, DMOD * DMOD);
  cvt(Wv, Wvb, DMOD * DMOD);  cvt(Wo, Wob, DMOD * DMOD);
  cvt(W1, W1b, DMOD * FMOD);  cvt(W2, W2b, FMOD * DMOD);

  // QKV projections (heads layouts)
  gemm_kernel<2><<<dim3(DMOD / 64, BL / 128), 256, 0, stream>>>(Xbf, Wqb, bq, Qb,   BL, DMOD, DMOD, 0);
  gemm_kernel<2><<<dim3(DMOD / 64, BL / 128), 256, 0, stream>>>(Xbf, Wkb, bk, Kbuf, BL, DMOD, DMOD, 0);
  gemm_kernel<3><<<dim3(DMOD / 64, BL / 128), 256, 0, stream>>>(Xbf, Wvb, bv, Vbuf, BL, DMOD, DMOD, 0);

  // flash attention with rel-pos bias; LDS = head table + 8 P bounce tiles
  const int attn_lds = TBL_BYTES_AL + 8 * 16 * 40 * 2;     // 171,856 B (< 320 KB WGP LDS)
  attn_kernel<<<dim3(LSEQ / 128, NB * NH), 256, attn_lds, stream>>>(
      Qb, Kbuf, Vbuf, pos, pos_mask, att_mask, padding, ptab, ctxb);

  // output projection + residual + LN1
  gemm_kernel<0><<<dim3(DMOD / 64, BL / 128), 256, 0, stream>>>(ctxb, Wob, bo, aout, BL, DMOD, DMOD, 0);
  ln_kernel<<<BL, 256, 0, stream>>>(tokens, aout, ln1g, ln1b, tok1, tok1b);

  // FFN (ReLU fused in first GEMM epilogue) + residual + LN2 -> d_out
  gemm_kernel<1><<<dim3(FMOD / 64, BL / 128), 256, 0, stream>>>(tok1b, W1b, b1, ff1b, BL, FMOD, DMOD, 1);
  gemm_kernel<0><<<dim3(DMOD / 64, BL / 128), 256, 0, stream>>>(ff1b, W2b, b2, ff2,  BL, DMOD, FMOD, 0);
  ln_kernel<<<BL, 256, 0, stream>>>(tok1, ff2, ln2g, ln2b, (float*)d_out, (__bf16*)nullptr);
}
